// SFNN_60378650247483
// MI455X (gfx1250) — matrix operand order repaired and checked by reading, not compiled, not run on hardware
//
#include <hip/hip_runtime.h>
#include <hip/hip_bf16.h>

typedef __attribute__((ext_vector_type(16))) _Float16 v16h;
typedef __attribute__((ext_vector_type(8)))  float    v8f;

#define N_TOT 1024
#define IN_NN 128
#define HID_NN 768
#define OUT_NN 128
#define DD 16
#define RLOOP 8   // rows per wave strip

// Fast activations: v_exp_f32 + v_rcp_f32 (avoid the IEEE div_fixup sequence).
__device__ __forceinline__ float fastrcp(float x) { return __builtin_amdgcn_rcpf(x); }
__device__ __forceinline__ float fsig(float x)  { return fastrcp(1.0f + __expf(-x)); }
__device__ __forceinline__ float ftanh(float x) { return 1.0f - 2.0f * fastrcp(1.0f + __expf(2.0f * x)); }

// ---------------------------------------------------------------------------
// Synapse pass: for rows [r0, r0 + nstrips*RLOOP), all 1024 columns, run the
// GRU via WMMA f16 (f32 accum), update hidden in-place, and write per-strip
// partial column sums of out = adj ? h_new*pre : 0 into Ppart[strip][c][d].
// One wave = 16 columns x RLOOP rows. 6 v_wmma_f32_16x16x32_f16 per row;
// r/z gates chain gh->gi through the C operand (D->C accumulate fast path).
// ---------------------------------------------------------------------------
__global__ __launch_bounds__(128) void synapse_kernel(
    const float* __restrict__ postbuf,        // N*16 current post values
    float* __restrict__ hidden,               // N*N*16 (workspace copy)
    float* __restrict__ Ppart,                // nstrips * N * 16 partial sums
    const unsigned char* __restrict__ adj,    // N*N bool (1 byte)
    const float* __restrict__ Wih,            // 48 x 33
    const float* __restrict__ Whh,            // 48 x 16
    const float* __restrict__ bih,            // 48
    const float* __restrict__ bhh,            // 48
    const float* __restrict__ reward_p,
    const float* __restrict__ lr_p,
    int r0)
{
    const int lane = threadIdx.x & 31;
    const int wave = blockIdx.x * (blockDim.x >> 5) + (threadIdx.x >> 5);
    const int cg   = wave & 63;          // 64 column-groups of 16
    const int rs   = wave >> 6;          // row strip
    const int c0   = cg * 16;
    const int rbase = r0 + rs * RLOOP;
    const int nl = lane & 15;            // column-in-tile / gate-dim index
    const int hs = lane >> 4;            // half select

    const float reward = reward_p[0];
    const float lr     = lr_p[0];

    // Per-lane gate biases; fold reward * Wih[:,32] into the input-gate bias.
    // r/z gates chain both WMMAs into one accumulator, so combine both biases.
    const float bR2 = bih[nl]      + bhh[nl]      + reward * Wih[(nl)      * 33 + 32];
    const float bZ2 = bih[16 + nl] + bhh[16 + nl] + reward * Wih[(16 + nl) * 33 + 32];
    const float biN = bih[32 + nl]                + reward * Wih[(32 + nl) * 33 + 32];
    const float bhN = bhh[32 + nl];

    // B tiles (K x 16). B[k][n] = W[n_tile*16 + n][k].
    // Lane layout: column n = nl, K = hs*8 + kk (kk<8), 16 + hs*8 + (kk-8) (kk>=8).
    v16h BihT[3], BhhT[3];
#pragma unroll
    for (int t = 0; t < 3; ++t) {
#pragma unroll
        for (int kk = 0; kk < 16; ++kk) {
            BihT[t][kk] = (_Float16)Wih[(16 * t + nl) * 33 + hs * 8 + kk + (kk & 8)];
            BhhT[t][kk] = (kk < 8) ? (_Float16)Whh[(16 * t + nl) * 16 + hs * 8 + kk]
                                   : (_Float16)0.0f;
        }
    }

    // X A-tile (16 synapses x K=32). Lane = row m=nl.
    // elements 0..7 : K = hs*8+e        -> pre[r][hs*8+e]      (per-row, loop varying)
    // elements 8..15: K = 16+hs*8+(e-8) -> post[c0+nl][hs*8+e] (loop invariant)
    v16h Xa;
#pragma unroll
    for (int e = 0; e < 8; ++e)
        Xa[8 + e] = (_Float16)postbuf[(c0 + nl) * DD + hs * 8 + e];

    v8f accOut = {};   // per-lane partial column sums (8 columns each)

    for (int rr = 0; rr < RLOOP; ++rr) {
        const int r = rbase + rr;
        const float* prerow = postbuf + (size_t)r * DD;

#pragma unroll
        for (int e = 0; e < 8; ++e)
            Xa[e] = (_Float16)prerow[hs * 8 + e];

        // H A-tile (16 x 16, K padded to 32 with zeros)
        const float* hrow = hidden + ((size_t)r * N_TOT + c0) * DD;
        v16h Ha;
#pragma unroll
        for (int e = 0; e < 8; ++e) {
            Ha[e]     = (_Float16)hrow[nl * DD + hs * 8 + e];
            Ha[8 + e] = (_Float16)0.0f;
        }

        // Bias-initialized accumulators (bias is per-column => per-lane const)
        v8f cR, cZ, cN, cHN;
#pragma unroll
        for (int i = 0; i < 8; ++i) {
            cR[i] = bR2; cZ[i] = bZ2; cN[i] = biN; cHN[i] = bhN;
        }

        // r gate: (x@WihR + h@WhhR + biases) in ONE chained accumulator
        cR  = __builtin_amdgcn_wmma_f32_16x16x32_f16(false, Ha, false, BhhT[0], (short)0, cR,  false, false);
        cR  = __builtin_amdgcn_wmma_f32_16x16x32_f16(false, Xa, false, BihT[0], (short)0, cR,  false, false);
        // z gate
        cZ  = __builtin_amdgcn_wmma_f32_16x16x32_f16(false, Ha, false, BhhT[1], (short)0, cZ,  false, false);
        cZ  = __builtin_amdgcn_wmma_f32_16x16x32_f16(false, Xa, false, BihT[1], (short)0, cZ,  false, false);
        // n gate needs gi and gh separately (r multiplies gh_n only)
        cN  = __builtin_amdgcn_wmma_f32_16x16x32_f16(false, Xa, false, BihT[2], (short)0, cN,  false, false);
        cHN = __builtin_amdgcn_wmma_f32_16x16x32_f16(false, Ha, false, BhhT[2], (short)0, cHN, false, false);

        const float pre_d = prerow[nl];

        // C/D layout: lane holds column d=nl for synapses m = hs*8 + i
#pragma unroll
        for (int i = 0; i < 8; ++i) {
            const int m = hs * 8 + i;
            const int c = c0 + m;
            float* hptr = hidden + ((size_t)r * N_TOT + c) * DD + nl;
            const float h  = *hptr;
            const float rg = fsig(cR[i]);
            const float zg = fsig(cZ[i]);
            const float ng = ftanh(cN[i] + rg * cHN[i]);
            const float g  = (1.0f - zg) * ng + zg * h;
            const bool  a  = adj[(size_t)r * N_TOT + c] != 0;
            const float hnew = a ? (h + g * lr) : h;
            *hptr = hnew;
            accOut[i] += a ? (hnew * pre_d) : 0.0f;
        }
    }

    // Deterministic partial-sum write (no atomics)
    float* Pp = Ppart + (size_t)rs * N_TOT * DD;
#pragma unroll
    for (int i = 0; i < 8; ++i) {
        const int c = c0 + hs * 8 + i;
        Pp[c * DD + nl] = accOut[i];
    }
}

// Fixed-order reduction of per-strip partials into a block column-sum array.
__global__ void reduce_partials(const float* __restrict__ P,
                                float* __restrict__ S, int nstrips)
{
    int idx = blockIdx.x * blockDim.x + threadIdx.x;
    if (idx >= N_TOT * DD) return;
    float a = 0.0f;
    for (int s = 0; s < nstrips; ++s)
        a += P[(size_t)s * N_TOT * DD + idx];
    S[idx] = a;
}

__global__ void zero_kernel(float* p, int n)
{
    int i = blockIdx.x * blockDim.x + threadIdx.x;
    if (i < n) p[i] = 0.0f;
}

// post[i,:] = tanh(obs[i] * rowsum(W_inp) + b_inp) for input rows
__global__ void post_input_kernel(const float* __restrict__ obs,
                                  const float* __restrict__ W,
                                  const float* __restrict__ b,
                                  float* __restrict__ postbuf)
{
    int idx = blockIdx.x * blockDim.x + threadIdx.x;
    if (idx >= IN_NN * DD) return;
    int i = idx / DD, dout = idx % DD;
    float rs = 0.0f;
    for (int k = 0; k < DD; ++k) rs += W[dout * DD + k];
    postbuf[i * DD + dout] = ftanh(obs[i] * rs + b[dout]);
}

// post[c,:] = tanh((S0+S1+S2)[c,:] @ W.T + b) for c in [c0, c0+cnt)
__global__ void post_act_kernel(const float* __restrict__ S0,
                                const float* __restrict__ S1,
                                const float* __restrict__ S2,
                                const float* __restrict__ W,
                                const float* __restrict__ b,
                                float* __restrict__ postbuf,
                                int c0, int cnt)
{
    int idx = blockIdx.x * blockDim.x + threadIdx.x;
    if (idx >= cnt * DD) return;
    int ci = idx / DD, dout = idx % DD;
    int c = c0 + ci;
    float acc = b[dout];
    for (int d = 0; d < DD; ++d) {
        float in = S0[c * DD + d] + S1[c * DD + d] + S2[c * DD + d];
        acc += in * W[dout * DD + d];
    }
    postbuf[c * DD + dout] = ftanh(acc);
}

// scores = post[N-OUT_N:, 0]; out = [argmax (first max), scores...]
__global__ void finalize_kernel(const float* __restrict__ postbuf,
                                float* __restrict__ out, int out_size)
{
    __shared__ float sv[OUT_NN];
    __shared__ int   si[OUT_NN];
    int j = threadIdx.x;
    float sc = postbuf[(size_t)(N_TOT - OUT_NN + j) * DD + 0];
    int base = (out_size >= OUT_NN + 1) ? 1 : 0;
    out[base + j] = sc;
    sv[j] = sc; si[j] = j;
    __syncthreads();
    for (int s = OUT_NN / 2; s > 0; s >>= 1) {
        if (j < s) {
            if (sv[j + s] > sv[j] || (sv[j + s] == sv[j] && si[j + s] < si[j])) {
                sv[j] = sv[j + s]; si[j] = si[j + s];
            }
        }
        __syncthreads();
    }
    if (j == 0 && out_size >= OUT_NN + 1) out[0] = (float)si[0];
}

extern "C" void kernel_launch(void* const* d_in, const int* in_sizes, int n_in,
                              void* d_out, int out_size, void* d_ws, size_t ws_size,
                              hipStream_t stream)
{
    (void)in_sizes; (void)n_in; (void)ws_size;
    const float*         obs     = (const float*)d_in[0];
    const float*         reward  = (const float*)d_in[1];
    const unsigned char* adj     = (const unsigned char*)d_in[2];  // jax bool = 1B
    const float*         hidden0 = (const float*)d_in[3];
    const float*         post0   = (const float*)d_in[4];
    const float*         lr      = (const float*)d_in[5];
    const float* W_inp   = (const float*)d_in[6];
    const float* b_inp   = (const float*)d_in[7];
    const float* Wih_inp = (const float*)d_in[8];
    const float* Whh_inp = (const float*)d_in[9];
    const float* bih_inp = (const float*)d_in[10];
    const float* bhh_inp = (const float*)d_in[11];
    const float* W_hid   = (const float*)d_in[12];
    const float* b_hid   = (const float*)d_in[13];
    const float* Wih_hid = (const float*)d_in[14];
    const float* Whh_hid = (const float*)d_in[15];
    const float* bih_hid = (const float*)d_in[16];
    const float* bhh_hid = (const float*)d_in[17];
    const float* W_out   = (const float*)d_in[18];
    const float* b_out   = (const float*)d_in[19];
    const float* Wih_out = (const float*)d_in[20];
    const float* Whh_out = (const float*)d_in[21];
    const float* bih_out = (const float*)d_in[22];
    const float* bhh_out = (const float*)d_in[23];

    // Workspace layout (floats)
    float* hidden  = (float*)d_ws;                          // N*N*16
    float* postbuf = hidden + (size_t)N_TOT * N_TOT * DD;   // N*16
    float* S0      = postbuf + N_TOT * DD;                  // N*16 (inp block)
    float* S1      = S0 + N_TOT * DD;                       // N*16 (hid block)
    float* S2      = S1 + N_TOT * DD;                       // N*16 (out block)
    float* Ppart   = S2 + N_TOT * DD;                       // up to 96*N*16

    hipMemcpyAsync(hidden, hidden0, (size_t)N_TOT * N_TOT * DD * sizeof(float),
                   hipMemcpyDeviceToDevice, stream);
    hipMemcpyAsync(postbuf, post0, (size_t)N_TOT * DD * sizeof(float),
                   hipMemcpyDeviceToDevice, stream);
    zero_kernel<<<(3 * N_TOT * DD + 255) / 256, 256, 0, stream>>>(S0, 3 * N_TOT * DD);

    const int SD_GRID = (N_TOT * DD + 127) / 128;
    for (int tick = 0; tick < 2; ++tick) {
        // Phase 1: input rows
        post_input_kernel<<<(IN_NN * DD + 127) / 128, 128, 0, stream>>>(obs, W_inp, b_inp, postbuf);
        {
            const int strips = IN_NN / RLOOP;               // 16
            synapse_kernel<<<strips * 16, 128, 0, stream>>>(postbuf, hidden, Ppart, adj,
                Wih_inp, Whh_inp, bih_inp, bhh_inp, reward, lr, 0);
            reduce_partials<<<SD_GRID, 128, 0, stream>>>(Ppart, S0, strips);
        }
        // Phase 2: hidden rows
        post_act_kernel<<<(HID_NN * DD + 127) / 128, 128, 0, stream>>>(
            S0, S1, S2, W_hid, b_hid, postbuf, IN_NN, HID_NN);
        {
            const int strips = HID_NN / RLOOP;              // 96
            synapse_kernel<<<strips * 16, 128, 0, stream>>>(postbuf, hidden, Ppart, adj,
                Wih_hid, Whh_hid, bih_hid, bhh_hid, reward, lr, IN_NN);
            reduce_partials<<<SD_GRID, 128, 0, stream>>>(Ppart, S1, strips);
        }
        // Phase 3: output rows
        post_act_kernel<<<(OUT_NN * DD + 127) / 128, 128, 0, stream>>>(
            S0, S1, S2, W_out, b_out, postbuf, N_TOT - OUT_NN, OUT_NN);
        if (tick == 0) {   // tick 1's out-synapse pass is unobservable in d_out
            const int strips = OUT_NN / RLOOP;              // 16
            synapse_kernel<<<strips * 16, 128, 0, stream>>>(postbuf, hidden, Ppart, adj,
                Wih_out, Whh_out, bih_out, bhh_out, reward, lr, N_TOT - OUT_NN);
            reduce_partials<<<SD_GRID, 128, 0, stream>>>(Ppart, S2, strips);
        }
    }

    finalize_kernel<<<1, OUT_NN, 0, stream>>>(postbuf, (float*)d_out, out_size);
}
